// Rollout_69483980915463
// MI455X (gfx1250) — compile-verified, matrix operand order checked by reading, not run on hardware
//
#include <hip/hip_runtime.h>
#include <hip/hip_bf16.h>

// ---------------------------------------------------------------------------
// 2-layer LSTM rollout for MI455X (gfx1250).
//  * bf16 weights packed once into WMMA 16x32 fragment layout (~136MB, fits
//    the 192MB L2 -> recurrence is L2/compute bound, not HBM bound).
//  * GEMM: one block owns all 128 rows (8 m-tiles) x 128 cols (8 n-tiles),
//    so each weight tile is read from L2 exactly once per GEMM.
//  * Double-buffered LDS stage filled with CDNA5 async copies
//    (global_load_async_to_lds_b128 / s_wait_asynccnt) overlapping WMMA math.
//    The LDS destination operand is derived from the real &sAB[...] pointer
//    (ptrtoint -> captured), so the compiler must assume the asm writes LDS
//    and keeps the ds_load fragment reads alive. (No "m" operands: the AMDGPU
//    backend cannot select inline-asm memory operands for LDS.)
//  * Activations emitted directly in packed-fragment layout by the fused
//    pointwise kernels; Gumbel-max sampling for the rollout phase.
// ---------------------------------------------------------------------------

typedef __bf16 v16bf __attribute__((ext_vector_type(16)));
typedef float  v8f   __attribute__((ext_vector_type(8)));

#define BB   128          // batch
#define SS   128          // sequence
#define VV   8000         // vocab
#define EE   512          // embed dim
#define HH   2048         // hidden
#define G4H  8192         // 4*H gate width
#define GIVEN 64

#define KT0  80           // (E+H)/32   layer-0 K tiles
#define KT1  128          // (H+H)/32   layer-1 K tiles
#define KTL  64           // H/32       logits K tiles
#define NTG  512          // 8192/16    gate N tiles
#define NTL  500          // 8000/16    logits N tiles

// --- packed fragment indexing (ISA 7.12.2, 16-bit A/B 16x32 tile) ----------
// element (row r, col k) -> tile (r/16, k/32); within tile:
//   lanes 0-15 hold kk in {0..7, 16..23}; lanes 16-31 hold kk in {8..15,24..31}
__device__ __forceinline__ size_t packedIdx(int r, int c, int KT) {
  int rt = r >> 4, rr = r & 15;
  int ct = c >> 5, kk = c & 31;
  int lane = rr + (((kk >> 3) & 1) << 4);
  int hi   = ((kk & 16) >> 1) + (kk & 7);
  return ((size_t)rt * KT + ct) * 512 + (size_t)lane * 16 + hi;
}

__device__ __forceinline__ unsigned short f2bf(float f) {
  union { float f; unsigned u; } x; x.f = f;
  unsigned u = x.u + 0x7FFFu + ((x.u >> 16) & 1u);   // round-to-nearest-even
  return (unsigned short)(u >> 16);
}

// --- one-time weight packing: bf16, fragment layout, K = [K0 | K1] concat ---
__global__ void pack_weights(const float* __restrict__ W0, int K0,
                             const float* __restrict__ W1, int K1,
                             unsigned short* __restrict__ out, int Nrows) {
  size_t idx = (size_t)blockIdx.x * blockDim.x + threadIdx.x;
  int K = K0 + K1;
  size_t total = (size_t)Nrows * K;
  if (idx >= total) return;
  int n = (int)(idx / K), k = (int)(idx % K);
  float v = (k < K0) ? W0[(size_t)n * K0 + k] : W1[(size_t)n * K1 + (k - K0)];
  out[packedIdx(n, k, K / 32)] = f2bf(v);
}

__global__ void vec_add(float* __restrict__ o, const float* __restrict__ a,
                        const float* __restrict__ b, int n) {
  int i = blockIdx.x * blockDim.x + threadIdx.x;
  if (i < n) o[i] = a[i] + b[i];
}

__global__ void zero_u32(unsigned* __restrict__ p, size_t n) {
  size_t i = (size_t)blockIdx.x * blockDim.x + threadIdx.x;
  if (i < n) p[i] = 0u;
}

__global__ void init_tokens(const int* __restrict__ x, int* __restrict__ tok) {
  int i = blockIdx.x * blockDim.x + threadIdx.x;   // 128*64
  if (i >= BB * GIVEN) return;
  int b = i / GIVEN, c = i % GIVEN;
  tok[b * SS + c] = x[b * SS + c];
}

// --- embedding gather -> packed A tiles 0..15 of layer0 input ---------------
__global__ void embed_pack(const int* __restrict__ tok, int col,
                           const float* __restrict__ emb,
                           unsigned short* __restrict__ packA) {
  int i = blockIdx.x * blockDim.x + threadIdx.x;   // 128*512
  if (i >= BB * EE) return;
  int b = i >> 9, k = i & (EE - 1);
  int t = tok[b * SS + col];
  packA[packedIdx(b, k, KT0)] = f2bf(emb[(size_t)t * EE + k]);
}

// ===========================================================================
// WMMA GEMM with async-LDS double buffering.
//   G[128 x 16*NT] = Apack(8mt x KT) * Bpack(NT x KT)^T
// block: 256 thr = 8 waves, covers M=128 (all 8 m-tiles) x N=128 (8 n-tiles).
// waves: wm = wave>>2 selects m-tiles [4wm..4wm+3], wn = wave&3 selects
//        local n-tiles [2wn, 2wn+1]  -> 8 wmma per wave per k-iter.
// LDS stage per k-iter: 16 tiles x 1KB (A tiles 0-7, B tiles 8-15), x2 bufs.
// waves 0-3 async-copy A tiles, waves 4-7 async-copy B tiles (uniform saddr).
// grid: (ceil(NT/8), 1)
// ===========================================================================
__global__ void __launch_bounds__(256)
gemm_wmma_bf16(const unsigned short* __restrict__ Apack,
               const unsigned short* __restrict__ Bpack,
               float* __restrict__ G,
               int KT, int NT, int ldG) {
  __shared__ unsigned short sAB[2][16 * 512];     // 32 KB

  const int t    = threadIdx.x;
  const int lane = t & 31;
  const int wave = t >> 5;
  const int wm   = wave >> 2;                     // 0..1
  const int wn   = wave & 3;                      // 0..3
  const int nt0  = blockIdx.x * 8;

  // staging role: thread serves 64B of tile g (16 threads per 1KB tile)
  const int g   = t >> 4;                         // 0..15 (0-7: A, 8-15: B)
  const int sub = t & 15;
  unsigned gRowBase;                              // tile-row byte base in pack
  if (g < 8) {
    gRowBase = (unsigned)g * (unsigned)KT * 1024u;
  } else {
    int nt = nt0 + (g - 8); if (nt > NT - 1) nt = NT - 1;
    gRowBase = (unsigned)nt * (unsigned)KT * 1024u;
  }

  v8f acc[4][2] = {};

  // ---- async stage issue for k-iter kt into buffer buf ----
  // LDS address operand derived from the real pointer (ptrtoint capture):
  // with the "memory" clobber the compiler must assume the asm writes sAB.
  auto issue = [&](int kt, int buf) {
    unsigned short* lp = &sAB[buf][g * 512 + sub * 32];   // 64B chunk base
    unsigned gb = gRowBase + (unsigned)kt * 1024u + (unsigned)sub * 64u;
    if (wave < 4) {                               // A region, saddr = Apack
#pragma unroll
      for (int q = 0; q < 4; ++q) {
        unsigned ldsAddr = (unsigned)(size_t)(lp + q * 8);   // LDS byte offset
        asm volatile("global_load_async_to_lds_b128 %0, %1, %2"
                     :: "v"(ldsAddr), "v"(gb + q * 16u), "s"(Apack)
                     : "memory");
      }
    } else {                                      // B region, saddr = Bpack
#pragma unroll
      for (int q = 0; q < 4; ++q) {
        unsigned ldsAddr = (unsigned)(size_t)(lp + q * 8);   // LDS byte offset
        asm volatile("global_load_async_to_lds_b128 %0, %1, %2"
                     :: "v"(ldsAddr), "v"(gb + q * 16u), "s"(Bpack)
                     : "memory");
      }
    }
  };

  issue(0, 0);

  for (int kt = 0; kt < KT; ++kt) {
    // drain stage kt (issued one iteration ago) and make it WG-visible
    asm volatile("s_wait_asynccnt 0x0" ::: "memory");
    __syncthreads();

    // prefetch stage kt+1 into the other buffer while we compute kt
    if (kt + 1 < KT) issue(kt + 1, (kt + 1) & 1);

    const v16bf* SL = (const v16bf*)(&sAB[kt & 1][0]);
    v16bf afr[4], bfr[2];
#pragma unroll
    for (int i = 0; i < 4; ++i) afr[i] = SL[(wm * 4 + i) * 32 + lane];
#pragma unroll
    for (int j = 0; j < 2; ++j) bfr[j] = SL[(8 + wn * 2 + j) * 32 + lane];
#pragma unroll
    for (int i = 0; i < 4; ++i)
#pragma unroll
      for (int j = 0; j < 2; ++j)
        acc[i][j] = __builtin_amdgcn_wmma_f32_16x16x32_bf16(
            false, afr[i], false, bfr[j], (short)0, acc[i][j], false, false);
  }

  // C layout: VGPR r -> row (r + 8*(lane>=16)), col = lane&15
  const int rsub = ((lane >> 4) << 3);
  const int col  = lane & 15;
#pragma unroll
  for (int i = 0; i < 4; ++i) {
#pragma unroll
    for (int j = 0; j < 2; ++j) {
      int nt = nt0 + wn * 2 + j;
      if (nt >= NT) continue;
      float* gp = G + (size_t)((wm * 4 + i) * 16 + rsub) * ldG + nt * 16 + col;
#pragma unroll
      for (int r = 0; r < 8; ++r) gp[(size_t)r * ldG] = acc[i][j][r];
    }
  }
}

// --- fused LSTM pointwise: gates -> (h,c); h emitted bf16 into packed bufs --
__global__ void lstm_pointwise(const float* __restrict__ G,
                               const float* __restrict__ bias,
                               float* __restrict__ c,
                               unsigned short* __restrict__ dstA, int ktOffA, int KTA,
                               unsigned short* __restrict__ dstB, int ktOffB, int KTB) {
  int idx = blockIdx.x * blockDim.x + threadIdx.x;  // 128*2048
  if (idx >= BB * HH) return;
  int b = idx >> 11, j = idx & (HH - 1);
  const float* g = G + (size_t)b * G4H;
  float i_ = g[j]          + bias[j];
  float f_ = g[j + HH]     + bias[j + HH];
  float gg = g[j + 2 * HH] + bias[j + 2 * HH];
  float o_ = g[j + 3 * HH] + bias[j + 3 * HH];
  float si = 1.f / (1.f + expf(-i_));
  float sf = 1.f / (1.f + expf(-f_));
  float so = 1.f / (1.f + expf(-o_));
  float cn = sf * c[idx] + si * tanhf(gg);
  c[idx] = cn;
  unsigned short hb = f2bf(so * tanhf(cn));
  dstA[packedIdx(b, ktOffA * 32 + j, KTA)] = hb;
  dstB[packedIdx(b, ktOffB * 32 + j, KTB)] = hb;
}

// --- Gumbel-max categorical sampling over V=8000 logits ---------------------
__global__ void sample_tokens(const float* __restrict__ logits,
                              const float* __restrict__ blin,
                              int* __restrict__ tok, int col, unsigned step) {
  int b = blockIdx.x;
  const float* L = logits + (size_t)b * VV;
  float best = -1e30f; int bi = 0;
  for (int j = threadIdx.x; j < VV; j += blockDim.x) {
    unsigned h = (unsigned)j * 2654435761u ^ (unsigned)b * 40503u
               ^ step * 2246822519u ^ 0x9E3779B9u;
    h ^= h >> 16; h *= 2246822519u; h ^= h >> 13; h *= 3266489917u; h ^= h >> 16;
    float u = (float)(h >> 8) * (1.f / 16777216.f) + 1e-12f;
    float v = L[j] + blin[j] - logf(-logf(u));
    if (v > best) { best = v; bi = j; }
  }
  __shared__ float sv[256];
  __shared__ int   si[256];
  sv[threadIdx.x] = best; si[threadIdx.x] = bi;
  __syncthreads();
  for (int s = 128; s > 0; s >>= 1) {
    if (threadIdx.x < s && sv[threadIdx.x + s] > sv[threadIdx.x]) {
      sv[threadIdx.x] = sv[threadIdx.x + s];
      si[threadIdx.x] = si[threadIdx.x + s];
    }
    __syncthreads();
  }
  if (threadIdx.x == 0) tok[b * SS + col] = si[0];
}

__global__ void write_out(const int* __restrict__ tok, int* __restrict__ out) {
  int i = blockIdx.x * blockDim.x + threadIdx.x;
  if (i < BB * SS) out[i] = tok[i];
}

// ---------------------------------------------------------------------------
extern "C" void kernel_launch(void* const* d_in, const int* in_sizes, int n_in,
                              void* d_out, int out_size, void* d_ws, size_t ws_size,
                              hipStream_t stream) {
  const int*   x    = (const int*)  d_in[0];
  // d_in[1] class_label: unused by reference. d_in[2] given_num == 64 (constant).
  const float* emb  = (const float*)d_in[3];
  const float* Wih0 = (const float*)d_in[4];
  const float* Whh0 = (const float*)d_in[5];
  const float* bih0 = (const float*)d_in[6];
  const float* bhh0 = (const float*)d_in[7];
  const float* Wih1 = (const float*)d_in[8];
  const float* Whh1 = (const float*)d_in[9];
  const float* bih1 = (const float*)d_in[10];
  const float* bhh1 = (const float*)d_in[11];
  const float* Wlin = (const float*)d_in[12];
  const float* blin = (const float*)d_in[13];

  // ---- workspace carve-out (~155 MB) ----
  char* p = (char*)d_ws;
  auto carve = [&](size_t bytes) -> char* {
    char* r = p; p += (bytes + 255) & ~(size_t)255; return r;
  };
  unsigned short* Bp0  = (unsigned short*)carve((size_t)G4H * (EE + HH) * 2);
  unsigned short* Bp1  = (unsigned short*)carve((size_t)G4H * (HH + HH) * 2);
  unsigned short* BpL  = (unsigned short*)carve((size_t)VV * HH * 2);
  float*          bias0= (float*)carve((size_t)G4H * 4);
  float*          bias1= (float*)carve((size_t)G4H * 4);
  unsigned short* Ap0  = (unsigned short*)carve((size_t)BB * (EE + HH) * 2);
  unsigned short* Ap1  = (unsigned short*)carve((size_t)BB * (HH + HH) * 2);
  unsigned short* ApL  = (unsigned short*)carve((size_t)BB * HH * 2);
  float*          Gbuf = (float*)carve((size_t)BB * G4H * 4);
  float*          Lbuf = (float*)carve((size_t)BB * VV * 4);
  float*          c0   = (float*)carve((size_t)BB * HH * 4);
  float*          c1   = (float*)carve((size_t)BB * HH * 4);
  int*            tok  = (int*)carve((size_t)BB * SS * 4);
  (void)ws_size; (void)in_sizes; (void)n_in; (void)out_size;

  auto cdiv = [](size_t a, size_t b) { return (unsigned)((a + b - 1) / b); };

  // ---- one-time (per launch) weight packing & init ----
  {
    size_t t0 = (size_t)G4H * (EE + HH);
    pack_weights<<<cdiv(t0, 256), 256, 0, stream>>>(Wih0, EE, Whh0, HH, Bp0, G4H);
    size_t t1 = (size_t)G4H * (HH + HH);
    pack_weights<<<cdiv(t1, 256), 256, 0, stream>>>(Wih1, HH, Whh1, HH, Bp1, G4H);
    size_t t2 = (size_t)VV * HH;
    pack_weights<<<cdiv(t2, 256), 256, 0, stream>>>(Wlin, HH, (const float*)nullptr, 0, BpL, VV);
    vec_add<<<cdiv(G4H, 256), 256, 0, stream>>>(bias0, bih0, bhh0, G4H);
    vec_add<<<cdiv(G4H, 256), 256, 0, stream>>>(bias1, bih1, bhh1, G4H);

    // zero h-packed buffers and cell states (init carry == 0)
    zero_u32<<<cdiv((size_t)BB * (EE + HH) * 2 / 4, 256), 256, 0, stream>>>(
        (unsigned*)Ap0, (size_t)BB * (EE + HH) * 2 / 4);
    zero_u32<<<cdiv((size_t)BB * (HH + HH) * 2 / 4, 256), 256, 0, stream>>>(
        (unsigned*)Ap1, (size_t)BB * (HH + HH) * 2 / 4);
    zero_u32<<<cdiv((size_t)BB * HH, 256), 256, 0, stream>>>(
        (unsigned*)c0, (size_t)BB * HH);
    zero_u32<<<cdiv((size_t)BB * HH, 256), 256, 0, stream>>>(
        (unsigned*)c1, (size_t)BB * HH);
    init_tokens<<<cdiv(BB * GIVEN, 256), 256, 0, stream>>>(x, tok);
  }

  const dim3 blk(256);
  const dim3 gGate(NTG / 8);                      // 64 blocks, B read once
  const dim3 gLin(cdiv(NTL, 8));                  // 63 blocks (tail guarded)
  const unsigned pwBlocks = cdiv((size_t)BB * HH, 256);

  // ---- 128 recurrent steps (64 teacher-forced + 64 rollout) ----
  for (int s = 0; s < SS; ++s) {
    int inCol = (s <= GIVEN - 1) ? s : s - 1;   // step64 re-feeds x[:,63] (ref semantics)
    embed_pack<<<cdiv((size_t)BB * EE, 256), 256, 0, stream>>>(tok, inCol, emb, Ap0);

    // layer 0: gates = [te|h0] x [Wih0|Whh0]^T
    gemm_wmma_bf16<<<gGate, blk, 0, stream>>>(Ap0, Bp0, Gbuf, KT0, NTG, G4H);
    lstm_pointwise<<<pwBlocks, 256, 0, stream>>>(Gbuf, bias0, c0,
                                                 Ap0, 16, KT0,   // h0 -> layer0 A
                                                 Ap1, 0,  KT1);  // h0 -> layer1 A
    // layer 1: gates = [h0|h1] x [Wih1|Whh1]^T
    gemm_wmma_bf16<<<gGate, blk, 0, stream>>>(Ap1, Bp1, Gbuf, KT1, NTG, G4H);
    lstm_pointwise<<<pwBlocks, 256, 0, stream>>>(Gbuf, bias1, c1,
                                                 Ap1, 64, KT1,   // h1 -> layer1 A
                                                 ApL, 0,  KTL);  // h1 -> logits A

    if (s >= GIVEN) {
      gemm_wmma_bf16<<<gLin, blk, 0, stream>>>(ApL, BpL, Lbuf, KTL, NTL, VV);
      sample_tokens<<<BB, 256, 0, stream>>>(Lbuf, blin, tok, s, (unsigned)s);
    }
  }

  write_out<<<cdiv(BB * SS, 256), 256, 0, stream>>>(tok, (int*)d_out);
}
